// RestrictedRotationalConv2D_66331474919797
// MI455X (gfx1250) — matrix operand order repaired and checked by reading, not compiled, not run on hardware
//
#include <hip/hip_runtime.h>
#include <math.h>

#define KH 7
#define KW 7
#define EH 5
#define EW 5
#define PADR 1
#define CIN 32
#define FOUT 64
#define HIN 64
#define WIN 64
#define BATCH 8
#define HO 58
#define WO 58
#define KDIM (EH*EW*CIN)          // 800
#define NPATCH (BATCH*HO*WO)      // 26912
#define TILE_M 16
#define NBLK (NPATCH/TILE_M)      // 1682 (exact)
#define ROWP (KDIM+8)             // 808 halves -> 1616B row stride, 16B aligned, de-conflicted

typedef __attribute__((ext_vector_type(16))) _Float16 v16h;
typedef __attribute__((ext_vector_type(8)))  _Float16 v8h;
typedef __attribute__((ext_vector_type(4)))  _Float16 v4h;
typedef __attribute__((ext_vector_type(8)))  float    v8f;

// ---- prep: intensity image S[b][h][w] = sum_c x ; W (f32) -> Wh (f16) ----
__global__ __launch_bounds__(256) void rr_prep_kernel(const float* __restrict__ x,
                                                      const float* __restrict__ W,
                                                      float* __restrict__ S,
                                                      _Float16* __restrict__ Wh) {
    int i = blockIdx.x * 256 + threadIdx.x;
    if (i < BATCH*HIN*WIN) {
        const float4* p = (const float4*)(x + (size_t)i * CIN);
        float s = 0.f;
#pragma unroll
        for (int j = 0; j < CIN/4; ++j) { float4 v = p[j]; s += v.x + v.y + v.z + v.w; }
        S[i] = s;
    }
    if (i < FOUT*KDIM) Wh[i] = (_Float16)W[i];
}

// ---- main: centroid/angle -> rotate -> WMMA GEMM (16 patches x 64 filters / block) ----
__global__ __launch_bounds__(128) void rr_main_kernel(const float* __restrict__ x,
                                                      const float* __restrict__ S,
                                                      const _Float16* __restrict__ Wh,
                                                      const float* __restrict__ bias,
                                                      float* __restrict__ out) {
    __shared__ _Float16 rotA[TILE_M][ROWP];
    __shared__ float    coef[TILE_M][6];
    __shared__ int      pinfo[TILE_M][3];

    const int tid   = threadIdx.x;
    const int pBase = blockIdx.x * TILE_M;

    // Phase 1: per-patch intensity centroid -> rotation affine coefficients
    if (tid < TILE_M) {
        int P  = pBase + tid;
        int wo = P % WO;
        int t  = P / WO;
        int ho = t % HO;
        int bb = t / HO;
        pinfo[tid][0] = bb; pinfo[tid][1] = ho; pinfo[tid][2] = wo;
        const float* Sp = S + ((size_t)bb * HIN + ho) * WIN + wo;
        float s0 = 0.f, sr = 0.f, sc = 0.f;
#pragma unroll
        for (int i = 0; i < KH; ++i)
#pragma unroll
            for (int j = 0; j < KW; ++j) {
                float v = Sp[i * WIN + j];
                s0 += v; sr += v * (float)i; sc += v * (float)j;
            }
        float denom = s0 + 1e-7f;
        float cr = sr / denom - 3.0f;
        float cc = sc / denom - 3.0f;
        float ang = atan2f(cr, cc + 1e-7f);
        float cs = cosf(ang), sn = sinf(ang);
        float inv = 1.0f / (1.0f + 1e-7f);
        float xo = (6.0f - (cs * 6.0f - sn * 6.0f)) * 0.5f;
        float yo = (6.0f - (sn * 6.0f + cs * 6.0f)) * 0.5f;
        coef[tid][0] =  cs * inv; coef[tid][1] = -sn * inv; coef[tid][2] = xo * inv;
        coef[tid][3] =  sn * inv; coef[tid][4] =  cs * inv; coef[tid][5] = yo * inv;
    }
    __syncthreads();

    // Phase 2: build rotated 5x5x32 vectors (f16) for 16 patches, cooperatively
    for (int idx = tid; idx < TILE_M * EH * EW; idx += 128) {
        int p   = idx / (EH * EW);
        int pos = idx - p * (EH * EW);
        int ey  = pos / EW;
        int ex  = pos - ey * EW;
        float fkx = (float)(ex + PADR);
        float fky = (float)(ey + PADR);
        float xin = coef[p][0] * fkx + coef[p][1] * fky + coef[p][2];
        float yin = coef[p][3] * fkx + coef[p][4] * fky + coef[p][5];
        float x0f = floorf(xin), y0f = floorf(yin);
        float wx1 = xin - x0f, wx0 = 1.f - wx1;
        float wy1 = yin - y0f, wy0 = 1.f - wy1;
        int x0 = (int)x0f, x1 = x0 + 1;
        int y0 = (int)y0f, y1 = y0 + 1;
        float vx0 = (x0 >= 0 && x0 < KW) ? 1.f : 0.f;
        float vx1 = (x1 >= 0 && x1 < KW) ? 1.f : 0.f;
        float vy0 = (y0 >= 0 && y0 < KH) ? 1.f : 0.f;
        float vy1 = (y1 >= 0 && y1 < KH) ? 1.f : 0.f;
        float w00 = wy0 * wx0 * vy0 * vx0;
        float w01 = wy0 * wx1 * vy0 * vx1;
        float w10 = wy1 * wx0 * vy1 * vx0;
        float w11 = wy1 * wx1 * vy1 * vx1;
        int cx0 = x0 < 0 ? 0 : (x0 > KW-1 ? KW-1 : x0);
        int cx1 = x1 < 0 ? 0 : (x1 > KW-1 ? KW-1 : x1);
        int cy0 = y0 < 0 ? 0 : (y0 > KH-1 ? KH-1 : y0);
        int cy1 = y1 < 0 ? 0 : (y1 > KH-1 ? KH-1 : y1);
        int bb = pinfo[p][0], ho = pinfo[p][1], wo = pinfo[p][2];
        const float* base = x + (((size_t)bb * HIN + ho) * WIN + wo) * CIN;
        const float4* q00 = (const float4*)(base + ((size_t)cy0 * WIN + cx0) * CIN);
        const float4* q01 = (const float4*)(base + ((size_t)cy0 * WIN + cx1) * CIN);
        const float4* q10 = (const float4*)(base + ((size_t)cy1 * WIN + cx0) * CIN);
        const float4* q11 = (const float4*)(base + ((size_t)cy1 * WIN + cx1) * CIN);
        _Float16* dst = &rotA[p][pos * CIN];
#pragma unroll
        for (int c4 = 0; c4 < CIN/4; ++c4) {
            float4 A = q00[c4], B = q01[c4], C = q10[c4], D = q11[c4];
            v4h h;
            h[0] = (_Float16)(w00 * A.x + w01 * B.x + w10 * C.x + w11 * D.x);
            h[1] = (_Float16)(w00 * A.y + w01 * B.y + w10 * C.y + w11 * D.y);
            h[2] = (_Float16)(w00 * A.z + w01 * B.z + w10 * C.z + w11 * D.z);
            h[3] = (_Float16)(w00 * A.w + w01 * B.w + w10 * C.w + w11 * D.w);
            *(v4h*)(dst + c4 * 4) = h;
        }
    }
    __syncthreads();

    // Phase 3: WMMA GEMM. Wave w handles filters [16w, 16w+16).
    const int lane  = tid & 31;
    const int wave  = tid >> 5;
    const int fBase = wave * 16;
    const int m     = lane & 15;     // row (A) / column (B) index
    const int hi    = lane >> 4;     // half-wave selector

    v8f acc = {};
    const _Float16* arow = &rotA[m][hi * 16];                        // A: K=hi*16..+15 per 32-block
    const _Float16* wrow = Wh + (size_t)(fBase + m) * KDIM + hi * 16; // B: K=hi*16..+15 per 32-block

#pragma unroll
    for (int kb = 0; kb < KDIM/32; ++kb) {
        const int k0 = kb * 32;
        v8h alo = *(const v8h*)(arow + k0);        // K = k0+16*hi .. +7
        v8h ahi = *(const v8h*)(arow + k0 + 8);    // K = k0+16*hi+8 .. +7
        v16h a  = __builtin_shufflevector(alo, ahi, 0,1,2,3,4,5,6,7,8,9,10,11,12,13,14,15);
        v16h b  = *(const v16h*)(wrow + k0);       // 32B contiguous, L2-hot
        acc = __builtin_amdgcn_wmma_f32_16x16x32_f16(false, a, false, b,
                                                     (short)0, acc, false, false);
    }

    // Phase 4: scatter D per 16x16 f32 C/D layout (+bias)
    const int n  = lane & 15;
    const float bz = bias[fBase + n];
#pragma unroll
    for (int v = 0; v < 8; ++v) {
        int p = pBase + v + hi * 8;                // M = v + 8*hi
        out[(size_t)p * FOUT + fBase + n] = acc[v] + bz;
    }
}

extern "C" void kernel_launch(void* const* d_in, const int* in_sizes, int n_in,
                              void* d_out, int out_size, void* d_ws, size_t ws_size,
                              hipStream_t stream) {
    (void)in_sizes; (void)n_in; (void)out_size; (void)ws_size;
    const float* x  = (const float*)d_in[0];
    const float* W  = (const float*)d_in[1];
    const float* b  = (const float*)d_in[2];
    float* out = (float*)d_out;

    float*    S  = (float*)d_ws;                                             // 131072 B
    _Float16* Wh = (_Float16*)((char*)d_ws + (size_t)BATCH*HIN*WIN*sizeof(float)); // 102400 B

    rr_prep_kernel<<<200, 256, 0, stream>>>(x, W, S, Wh);
    rr_main_kernel<<<NBLK, 128, 0, stream>>>(x, S, Wh, b, out);
}
